// CossimBCE_79800492359795
// MI455X (gfx1250) — matrix operand order repaired and checked by reading, not compiled, not run on hardware
//
#include <hip/hip_runtime.h>
#include <hip/hip_bf16.h>
#include <math.h>

// Problem constants (from reference)
#define BQ   8
#define NQ   2048
#define MQ   2048
#define CQ   256
#define EPSQ 1e-8f

typedef _Float16 v8h  __attribute__((ext_vector_type(8)));
typedef _Float16 v16h __attribute__((ext_vector_type(16)));
typedef float    v8f  __attribute__((ext_vector_type(8)));
typedef unsigned int v4u __attribute__((ext_vector_type(4)));
typedef int      v4i  __attribute__((ext_vector_type(4)));
typedef int      v8i  __attribute__((ext_vector_type(8)));

#define HAVE_TDM __has_builtin(__builtin_amdgcn_tensor_load_to_lds)

// LDS B-panel row stride in halves: 32 data + 8 pad  -> 80 bytes (20 banks)
#define BSTRIDE 40

static __device__ __forceinline__ v16h pack16(v8h lo, v8h hi) {
    return __builtin_shufflevector(lo, hi, 0, 1, 2, 3, 4, 5, 6, 7,
                                            8, 9, 10, 11, 12, 13, 14, 15);
}

#if HAVE_TDM
// Issue a TDM load: 128 rows x 32 halves (row stride 256 halves) from global,
// padded +16B per 64B row into LDS (row stride 80B). 2D tensor descriptor.
static __device__ __forceinline__ void tdm_load_panel(unsigned int ldsAddr,
                                                      const _Float16* gptr) {
    unsigned long long ga = (unsigned long long)(uintptr_t)gptr;
    v4u g0;
    g0[0] = 1u;                                   // count=1, user mode, no gather
    g0[1] = ldsAddr;                              // lds_addr (bytes)
    g0[2] = (unsigned int)ga;                     // global_addr[31:0]
    g0[3] = (unsigned int)((ga >> 32) & 0x01FFFFFFu) | (2u << 30); // ga[56:32] | type=2

    v8i g1;
    g1[0] = (1 << 16)      // data_size = 2 bytes
          | (1 << 20)      // pad_enable
          | (3 << 22)      // pad_interval: 16 DWORDs (64B) between pads
          | (3 << 25);     // pad_amount: 4 DWORDs (16B)
    g1[1] = (int)(256u << 16);   // tensor_dim0 = 256 (bits 79:48, low half)
    g1[2] = (int)(2048u << 16);  // tensor_dim1 = 2048 (bits 111:80, low half)
    g1[3] = (int)(32u << 16);    // tile_dim0 = 32 (bits 127:112)
    g1[4] = 128;                 // tile_dim1 = 128 (bits 143:128)
    g1[5] = 256;                 // tensor_dim0_stride = 256 (bits 207:160)
    g1[6] = 0;
    g1[7] = 0;

    v4i g2 = {0, 0, 0, 0};
    v4i g3 = {0, 0, 0, 0};
#if __clang_major__ >= 23
    v8i g4 = {0, 0, 0, 0, 0, 0, 0, 0};
    __builtin_amdgcn_tensor_load_to_lds(g0, g1, g2, g3, g4, 0);
#else
    __builtin_amdgcn_tensor_load_to_lds(g0, g1, g2, g3, 0);
#endif
}
#endif

// ---------------------------------------------------------------------------
// Kernel 0: zero the per-batch accumulators in workspace
// ---------------------------------------------------------------------------
__global__ void cdna5_zero_acc(float* sAcc, float* cAcc, unsigned* negF, unsigned* posF) {
    int i = threadIdx.x;
    if (i < BQ) {
        sAcc[i] = 0.0f;
        cAcc[i] = 0.0f;
        negF[i] = 0u;
        posF[i] = 0u;
    }
}

// ---------------------------------------------------------------------------
// Kernel 1: L2-normalize rows of x1 and x2 (C=256) into f16 workspace.
// One wave per row; each lane handles 8 consecutive floats.
// ---------------------------------------------------------------------------
__global__ void cdna5_normalize(const float* __restrict__ x1,
                                const float* __restrict__ x2,
                                _Float16* __restrict__ n1h,
                                _Float16* __restrict__ n2h) {
    const int lane = threadIdx.x & 31;
    const int wv   = threadIdx.x >> 5;
    const int row  = blockIdx.x * 8 + wv;
    const int totalRows = BQ * NQ;

    const float* __restrict__ src;
    _Float16*    __restrict__ dst;
    if (row < totalRows) {
        src = x1 + (size_t)row * CQ;
        dst = n1h + (size_t)row * CQ;
    } else {
        src = x2 + (size_t)(row - totalRows) * CQ;
        dst = n2h + (size_t)(row - totalRows) * CQ;
    }

    const float4* p = (const float4*)(src) + (size_t)lane * 2;
    float4 a = p[0];
    float4 b = p[1];

    float ss = a.x * a.x + a.y * a.y + a.z * a.z + a.w * a.w +
               b.x * b.x + b.y * b.y + b.z * b.z + b.w * b.w;
#pragma unroll
    for (int m = 16; m >= 1; m >>= 1)
        ss += __shfl_xor(ss, m, 32);

    float scale = 1.0f / fmaxf(sqrtf(ss), EPSQ);

    v8h h;
    h[0] = (_Float16)(a.x * scale);
    h[1] = (_Float16)(a.y * scale);
    h[2] = (_Float16)(a.z * scale);
    h[3] = (_Float16)(a.w * scale);
    h[4] = (_Float16)(b.x * scale);
    h[5] = (_Float16)(b.y * scale);
    h[6] = (_Float16)(b.z * scale);
    h[7] = (_Float16)(b.w * scale);
    *(v8h*)(dst + (size_t)lane * 8) = h;
}

// ---------------------------------------------------------------------------
// Kernel 2: fused WMMA GEMM (cos_sim = n1 @ n2^T) + sigmoid-BCE epilogue.
// Grid: (M/128, N/128, B), block: 256 threads = 8 waves (wave32).
// B panel staged in LDS by the Tensor Data Mover, double-buffered per K-step.
// Each wave computes a 16x128 strip: 8 accumulator tiles, K=256 in 8 steps.
// ---------------------------------------------------------------------------
__global__ void __launch_bounds__(256)
cdna5_gemm_loss(const _Float16* __restrict__ n1h,
                const _Float16* __restrict__ n2h,
                const int* __restrict__ z,
                const float* __restrict__ tPtr,
                const float* __restrict__ bPtr,
                float* sAcc, float* cAcc,
                unsigned* negF, unsigned* posF) {
    const int lane   = threadIdx.x & 31;
    const int wv     = threadIdx.x >> 5;
    const int bz     = blockIdx.z;
    const int mBase  = blockIdx.x * 128;
    const int nBase  = blockIdx.y * 128 + wv * 16;

    const int hiHalf = lane >> 4;
    const int lane15 = lane & 15;
    const int kAoff  = hiHalf * 16;   // A 16x32 layout: contiguous 16 halves
    const int kBoff  = hiHalf * 16;   // B 32x16 layout: contiguous 16 halves

    const _Float16* __restrict__ Arow =
        n1h + ((size_t)bz * NQ + (size_t)(nBase + lane15)) * CQ;

    // Warm L2 for the z tile read in the epilogue.
    {
        const size_t zi = ((size_t)bz * NQ + (size_t)(nBase + hiHalf * 8)) * MQ
                          + (size_t)(mBase + lane15);
        __builtin_prefetch(z + zi, 0, 1);
    }

    v8f acc[8];
#pragma unroll
    for (int j = 0; j < 8; ++j)
#pragma unroll
        for (int r = 0; r < 8; ++r)
            acc[j][r] = 0.0f;

#if HAVE_TDM
    // Double-buffered LDS B panel: [2][128 rows][40 halves] (80B row stride).
    __shared__ _Float16 ldsB[2][128 * BSTRIDE];

    const _Float16* Bbase = n2h + ((size_t)bz * MQ + (size_t)mBase) * CQ;

    if (wv == 0) {
        tdm_load_panel((unsigned int)(uintptr_t)&ldsB[0][0], Bbase + 0);
        tdm_load_panel((unsigned int)(uintptr_t)&ldsB[1][0], Bbase + 32);
        __builtin_amdgcn_s_wait_tensorcnt((short)1);   // buffer 0 ready
    }
    __syncthreads();

#pragma unroll
    for (int i = 0; i < 8; ++i) {
        const int k0 = i * 32;

        v8h a0 = *(const v8h*)(Arow + k0 + kAoff);
        v8h a1 = *(const v8h*)(Arow + k0 + kAoff + 8);
        v16h afrag = pack16(a0, a1);

        const _Float16* lbuf = &ldsB[i & 1][0];
#pragma unroll
        for (int j = 0; j < 8; ++j) {
            const int roff = (j * 16 + lane15) * BSTRIDE + kBoff;
            v8h b0 = *(const v8h*)(lbuf + roff);
            v8h b1 = *(const v8h*)(lbuf + roff + 8);
            v16h bfrag = pack16(b0, b1);
            acc[j] = __builtin_amdgcn_wmma_f32_16x16x32_f16(
                false, afrag, false, bfrag, (short)0, acc[j], false, false);
        }

        if (i < 7) {
            __syncthreads();               // all waves done reading ldsB[i&1]
            if (wv == 0) {
                if (i + 2 < 8) {
                    tdm_load_panel((unsigned int)(uintptr_t)&ldsB[i & 1][0],
                                   Bbase + (k0 + 64));
                    __builtin_amdgcn_s_wait_tensorcnt((short)1);  // next buf ready
                } else {
                    __builtin_amdgcn_s_wait_tensorcnt((short)0);  // last buf ready
                }
            }
            __syncthreads();               // ldsB[(i+1)&1] ready for all waves
        }
    }
#else
    // Fallback: direct global loads of B fragments (round-1 path).
    const _Float16* __restrict__ Brow =
        n2h + ((size_t)bz * MQ + (size_t)(mBase + lane15)) * CQ;
#pragma unroll
    for (int k0 = 0; k0 < CQ; k0 += 32) {
        v8h a0 = *(const v8h*)(Arow + k0 + kAoff);
        v8h a1 = *(const v8h*)(Arow + k0 + kAoff + 8);
        v16h afrag = pack16(a0, a1);
#pragma unroll
        for (int j = 0; j < 8; ++j) {
            const _Float16* bp = Brow + (size_t)j * 16 * CQ + k0 + kBoff;
            v8h b0 = *(const v8h*)(bp);
            v8h b1 = *(const v8h*)(bp + 8);
            v16h bfrag = pack16(b0, b1);
            acc[j] = __builtin_amdgcn_wmma_f32_16x16x32_f16(
                false, afrag, false, bfrag, (short)0, acc[j], false, false);
        }
    }
#endif

    // ---------------- fused loss epilogue ----------------
    const float tv = tPtr[0];
    const float bv = bPtr[0];

    float s = 0.0f, c = 0.0f;
    float fNeg = 0.0f, fPos = 0.0f;

    const int mIdx = mBase + lane15;  // D layout: lane -> column (m)
    const int nOff = hiHalf * 8;      // D layout: VGPR r -> row (n = r + 8*hiHalf)

#pragma unroll
    for (int j = 0; j < 8; ++j) {
        const int m = mIdx + j * 16;
#pragma unroll
        for (int r = 0; r < 8; ++r) {
            const int n = nBase + nOff + r;
            const int zz = z[((size_t)bz * NQ + (size_t)n) * MQ + (size_t)m];
            if (zz != 0) {
                float y = (float)zz * (tv * acc[j][r] - bv);
                // stable log_sigmoid(y) = min(y,0) - log(1 + exp(-|y|))
                float e  = __expf(-fabsf(y));
                float ls = fminf(y, 0.0f) - __logf(1.0f + e);
                s += ls;
                c += 1.0f;
                if (zz < 0) fNeg = 1.0f;
                if (zz > 0) fPos = 1.0f;
            }
        }
    }

    // wave32 reduction
#pragma unroll
    for (int off = 16; off >= 1; off >>= 1) {
        s    += __shfl_down(s,    (unsigned)off, 32);
        c    += __shfl_down(c,    (unsigned)off, 32);
        fNeg += __shfl_down(fNeg, (unsigned)off, 32);
        fPos += __shfl_down(fPos, (unsigned)off, 32);
    }

    __shared__ float redS[8], redC[8], redN[8], redP[8];
    if (lane == 0) {
        redS[wv] = s;
        redC[wv] = c;
        redN[wv] = fNeg;
        redP[wv] = fPos;
    }
    __syncthreads();

    if (threadIdx.x == 0) {
        float S = 0.0f, C = 0.0f, Nf = 0.0f, Pf = 0.0f;
#pragma unroll
        for (int i = 0; i < 8; ++i) {
            S += redS[i];
            C += redC[i];
            Nf += redN[i];
            Pf += redP[i];
        }
        atomicAdd(&sAcc[bz], S);
        atomicAdd(&cAcc[bz], C);
        if (Nf > 0.0f) atomicOr(&negF[bz], 1u);
        if (Pf > 0.0f) atomicOr(&posF[bz], 1u);
    }
}

// ---------------------------------------------------------------------------
// Kernel 3: batch filter + final mean
// ---------------------------------------------------------------------------
__global__ void cdna5_finalize(const float* sAcc, const float* cAcc,
                               const unsigned* negF, const unsigned* posF,
                               float* out) {
    if (threadIdx.x == 0 && blockIdx.x == 0) {
        float S = 0.0f, C = 0.0f;
        for (int b = 0; b < BQ; ++b) {
            if (negF[b] && posF[b]) {
                S += sAcc[b];
                C += cAcc[b];
            }
        }
        out[0] = (C > 0.0f) ? (-S / C) : 0.0f;
    }
}

// ---------------------------------------------------------------------------
// Launcher
// ---------------------------------------------------------------------------
extern "C" void kernel_launch(void* const* d_in, const int* in_sizes, int n_in,
                              void* d_out, int out_size, void* d_ws, size_t ws_size,
                              hipStream_t stream) {
    const int*   z  = (const int*)d_in[0];     // [B,N,M] labels in {-1,0,1}
    const float* x1 = (const float*)d_in[1];   // [B,N,C]
    const float* x2 = (const float*)d_in[2];   // [B,M,C]
    const float* tP = (const float*)d_in[3];   // scalar
    const float* bP = (const float*)d_in[4];   // scalar
    float* out = (float*)d_out;

    (void)in_sizes; (void)n_in; (void)out_size; (void)ws_size;

    const size_t n1Bytes = (size_t)BQ * NQ * CQ * sizeof(_Float16);  // 8 MiB
    const size_t n2Bytes = (size_t)BQ * MQ * CQ * sizeof(_Float16);  // 8 MiB
    char* ws = (char*)d_ws;
    _Float16* n1h  = (_Float16*)(ws);
    _Float16* n2h  = (_Float16*)(ws + n1Bytes);
    float*    sAcc = (float*)(ws + n1Bytes + n2Bytes);
    float*    cAcc = sAcc + BQ;
    unsigned* negF = (unsigned*)(cAcc + BQ);
    unsigned* posF = negF + BQ;

    cdna5_zero_acc<<<1, 32, 0, stream>>>(sAcc, cAcc, negF, posF);

    cdna5_normalize<<<(2 * BQ * NQ) / 8, 256, 0, stream>>>(x1, x2, n1h, n2h);

    dim3 grid(MQ / 128, NQ / 128, BQ);  // (16, 16, 8)
    cdna5_gemm_loss<<<grid, 256, 0, stream>>>(n1h, n2h, z, tP, bP,
                                              sAcc, cAcc, negF, posF);

    cdna5_finalize<<<1, 32, 0, stream>>>(sAcc, cAcc, negF, posF, out);
}
